// BigbirdBlockSpareAttention_65979287601679
// MI455X (gfx1250) — matrix operand order repaired and checked by reading, not compiled, not run on hardware
//
#include <hip/hip_runtime.h>

// BigBird block-sparse attention for MI455X (gfx1250, wave32).
// Flash-attention over the per-row-block sparse key list using
// v_wmma_f32_16x16x32_bf16 (fp32 accumulate). K/V are converted to bf16
// once at staging time (K row-major, V transposed in LDS) so every WMMA
// B operand is two aligned ds_load_b128s. bf16 A operands are assembled
// as packed 32-bit words and bit_cast to the vector type (no sub-register
// moves). Softmax row stats use one shfl_xor(16) per reduction.

typedef __attribute__((ext_vector_type(16))) __bf16    v16bf;
typedef __attribute__((ext_vector_type(8)))  __bf16    v8bf;
typedef __attribute__((ext_vector_type(8)))  float     v8f;
typedef __attribute__((ext_vector_type(8)))  unsigned  v8u;

#define BB    2
#define HH    16
#define MM    4096
#define DD    64
#define WBLK  64
#define NBLK  (MM / WBLK)     // 64 blocks
#define RR    3
#define NEGV  (-10000.0f)
#define KC    32              // keys per streamed chunk
#define KSTR  72              // bf16 elems per K LDS row (144B: 16B-aligned, bank-staggered)
#define VSTR  40              // bf16 elems per V^T LDS row (80B: 16B-aligned)
#define PLD   36              // floats per S-scratch row (144B: 16B-aligned)

#define LDS_FENCE() asm volatile("" ::: "memory")

__device__ __forceinline__ unsigned bfb(float f) {       // fp32 -> bf16 bits (RNE)
  unsigned u = __builtin_bit_cast(unsigned, f);
  u += 0x7FFFu + ((u >> 16) & 1u);
  return u >> 16;
}
__device__ __forceinline__ unsigned pkbf(float a, float b) {  // two bf16 in one dword
#if __has_builtin(__builtin_amdgcn_cvt_pk_bf16_f32)
  return __builtin_bit_cast(unsigned, __builtin_amdgcn_cvt_pk_bf16_f32(a, b));
#else
  return bfb(a) | (bfb(b) << 16);
#endif
}
__device__ __forceinline__ __bf16 f2bf(float f) {
  unsigned short hs = (unsigned short)bfb(f);
  return __builtin_bit_cast(__bf16, hs);
}
__device__ __forceinline__ v16bf ld16bf(const __bf16* p, int hl) {  // 2x aligned b128
  v8bf lo = *(const v8bf*)(p + 8 * hl);
  v8bf hi = *(const v8bf*)(p + 16 + 8 * hl);
  return __builtin_shufflevector(lo, hi, 0,1,2,3,4,5,6,7,8,9,10,11,12,13,14,15);
}

__launch_bounds__(128)
__global__ void bigbird_sparse_attn(
    const float* __restrict__ q, const float* __restrict__ kk,
    const float* __restrict__ vv, const float* __restrict__ to_mask,
    const float* __restrict__ from_mask, const int* __restrict__ rand_attn,
    float* __restrict__ out)
{
  __shared__ __bf16 KldB[KC * KSTR];       // K chunk bf16, [key][d]
  __shared__ __bf16 VT[DD * VSTR];         // V chunk bf16, transposed [d][key]
  __shared__ float  Pld[4 * 16 * PLD];     // per-wave S scratch (C->A relayout)
  __shared__ float  Mr[4 * 16], Lr[4 * 16], Cr[4 * 16];  // per-wave row stats
  __shared__ float  Tm[KC];
  __shared__ int    kblocks[NBLK];
  __shared__ int    nkb_s;

  const int rb  = blockIdx.x, h = blockIdx.y, b = blockIdx.z;
  const int tid = threadIdx.x;
  const int lane = tid & 31, wave = tid >> 5;
  const int hl = lane >> 4, l15 = lane & 15;

  // ---- enumerate key blocks (BigBird pattern) ----
  if (tid == 0) {
    int n = 0;
    if (rb == 0 || rb == NBLK - 1) {
      for (int i = 0; i < NBLK; ++i) kblocks[n++] = i;
    } else {
      const int* ra = rand_attn + ((size_t)(b * HH + h) * (NBLK - 2) + (rb - 1)) * RR;
      if (rb == 1) { kblocks[0]=0; kblocks[1]=1; kblocks[2]=2; kblocks[3]=NBLK-1; n=4; }
      else if (rb == NBLK - 2) { kblocks[0]=0; kblocks[1]=NBLK-3; kblocks[2]=NBLK-2; kblocks[3]=NBLK-1; n=4; }
      else { kblocks[0]=0; kblocks[1]=rb-1; kblocks[2]=rb; kblocks[3]=rb+1; kblocks[4]=NBLK-1; n=5; }
      for (int j = 0; j < RR; ++j) {
        int t = ra[j];
        t = t < 0 ? 0 : (t > NBLK - 1 ? NBLK - 1 : t);
        kblocks[n++] = t;
      }
    }
    nkb_s = n;
  }
  if (lane < 16) { Mr[wave * 16 + lane] = -3.0e38f; Lr[wave * 16 + lane] = 0.0f; }
  __syncthreads();
  const int nkb = nkb_s;

  // ---- this wave's 16 Q rows into WMMA A-layout, packed word-wise ----
  const size_t bh = (size_t)(b * HH + h);
  const float* qbase = q + (bh * MM + (size_t)rb * WBLK + wave * 16) * DD;
  const float* qrow = qbase + l15 * DD;
  v16bf aq[2];
  for (int c = 0; c < 2; ++c) {
    v8u aw;
    for (int p = 0; p < 4; ++p) {              // elems 2p,2p+1  <- d,d+1
      int d = 32 * c + hl * 8 + 2 * p;
      aw[p] = pkbf(qrow[d], qrow[d + 1]);
    }
    for (int p = 0; p < 4; ++p) {              // elems 8+2p,9+2p <- d,d+1
      int d = 32 * c + 16 + hl * 8 + 2 * p;
      aw[4 + p] = pkbf(qrow[d], qrow[d + 1]);
    }
    aq[c] = __builtin_bit_cast(v16bf, aw);
  }

  v8f acc[4] = {};                  // 16x64 fp32 accum (C layout, 4 d-tiles)
  float* MrW = Mr + wave * 16;
  float* LrW = Lr + wave * 16;
  float* CrW = Cr + wave * 16;
  float* Pw  = Pld + wave * 16 * PLD;

  const float* kbh = kk + bh * MM * DD;
  const float* vbh = vv + bh * MM * DD;

  for (int kbi = 0; kbi < nkb; ++kbi) {
    const int kb = kblocks[kbi];
    for (int ch = 0; ch < 2; ++ch) {
      const int kstart = kb * WBLK + ch * KC;
      __syncthreads();   // previous chunk's LDS consumers done

      // ---- stage K (bf16 row-major) + V (bf16 transposed) + to_mask ----
      {
        const float* kg = kbh + (size_t)kstart * DD;
        const float* vg = vbh + (size_t)kstart * DD;
        for (int i = tid; i < KC * (DD / 4); i += 128) {
          int r = i >> 4, c4 = i & 15;
          float4 ka = *(const float4*)(kg + r * DD + c4 * 4);
          float4 va = *(const float4*)(vg + r * DD + c4 * 4);
          uint2 kp;
          kp.x = pkbf(ka.x, ka.y);
          kp.y = pkbf(ka.z, ka.w);
          *(uint2*)&KldB[r * KSTR + c4 * 4] = kp;   // 8B aligned (144B stride)
          VT[(c4 * 4 + 0) * VSTR + r] = f2bf(va.x);
          VT[(c4 * 4 + 1) * VSTR + r] = f2bf(va.y);
          VT[(c4 * 4 + 2) * VSTR + r] = f2bf(va.z);
          VT[(c4 * 4 + 3) * VSTR + r] = f2bf(va.w);
        }
        if (tid < KC) Tm[tid] = to_mask[(size_t)b * MM + kstart + tid];
        if (!(kbi == nkb - 1 && ch == 1)) {        // gfx1250 global_prefetch_b8
          int nb2 = (ch == 0) ? kb : kblocks[kbi + 1];
          int nc2 = (ch == 0) ? 1 : 0;
          const float* pf = kbh + (size_t)(nb2 * WBLK + nc2 * KC) * DD;
          __builtin_prefetch(pf + (size_t)tid * 16, 0, 1);
        }
      }
      __syncthreads();

      // ---- S = Q*K^T : 2 key-tiles x 2 d-chunks of bf16 WMMA ----
      v8f s0 = {}, s1 = {};
      for (int c = 0; c < 2; ++c) {
        v16bf bk0 = ld16bf(&KldB[l15 * KSTR + 32 * c], hl);        // keys 0..15
        v16bf bk1 = ld16bf(&KldB[(16 + l15) * KSTR + 32 * c], hl); // keys 16..31
        s0 = __builtin_amdgcn_wmma_f32_16x16x32_bf16(false, aq[c], false, bk0,
                                                     (short)0, s0, false, false);
        s1 = __builtin_amdgcn_wmma_f32_16x16x32_bf16(false, aq[c], false, bk1,
                                                     (short)0, s1, false, false);
      }

      // ---- scale + mask, dump S to LDS in C layout ----
      const float mk0 = (1.0f - Tm[l15]) * NEGV;
      const float mk1 = (1.0f - Tm[16 + l15]) * NEGV;
      for (int j = 0; j < 8; ++j) {
        Pw[(j + 8 * hl) * PLD + l15]      = s0[j] * 0.125f + mk0;
        Pw[(j + 8 * hl) * PLD + 16 + l15] = s1[j] * 0.125f + mk1;
      }
      LDS_FENCE();   // wave-private region: DS is in-order per wave

      // ---- softmax: lane owns row=l15, its 16 A-layout columns ----
      const float* Sr = Pw + l15 * PLD;
      float pv[16];
      {
        float4 a0 = *(const float4*)(Sr + hl * 8);
        float4 a1 = *(const float4*)(Sr + hl * 8 + 4);
        float4 a2 = *(const float4*)(Sr + 16 + hl * 8);
        float4 a3 = *(const float4*)(Sr + 16 + hl * 8 + 4);
        pv[0]=a0.x;  pv[1]=a0.y;  pv[2]=a0.z;  pv[3]=a0.w;
        pv[4]=a1.x;  pv[5]=a1.y;  pv[6]=a1.z;  pv[7]=a1.w;
        pv[8]=a2.x;  pv[9]=a2.y;  pv[10]=a2.z; pv[11]=a2.w;
        pv[12]=a3.x; pv[13]=a3.y; pv[14]=a3.z; pv[15]=a3.w;
      }
      float cmax = pv[0];
      for (int e = 1; e < 16; ++e) cmax = fmaxf(cmax, pv[e]);
      cmax = fmaxf(cmax, __shfl_xor(cmax, 16, 32));   // combine the two halves
      const float mold = MrW[l15];
      const float mnew = fmaxf(mold, cmax);
      const float corr = __expf(mold - mnew);
      float rs = 0.0f;
      v8u apw;
      for (int p = 0; p < 8; ++p) {
        float e0 = __expf(pv[2 * p]     - mnew);
        float e1 = __expf(pv[2 * p + 1] - mnew);
        rs += e0 + e1;
        apw[p] = pkbf(e0, e1);
      }
      v16bf ap = __builtin_bit_cast(v16bf, apw);
      rs += __shfl_xor(rs, 16, 32);
      if (hl == 0) {
        MrW[l15] = mnew;
        LrW[l15] = LrW[l15] * corr + rs;
        CrW[l15] = corr;
      }
      LDS_FENCE();

      // ---- rescale accumulators with this chunk's per-row correction ----
      for (int j = 0; j < 8; ++j) {
        float cj = CrW[j + 8 * hl];
        acc[0][j] *= cj; acc[1][j] *= cj; acc[2][j] *= cj; acc[3][j] *= cj;
      }

      // ---- O += P(16x32) * V(32x64): one WMMA per 16-col d-tile ----
      for (int t = 0; t < 4; ++t) {
        v16bf bv = ld16bf(&VT[(t * 16 + l15) * VSTR], hl);
        acc[t] = __builtin_amdgcn_wmma_f32_16x16x32_bf16(false, ap, false, bv,
                                                         (short)0, acc[t], false, false);
      }
    }
  }

  // ---- normalize + from_mask, store to (B, M, H, D) ----
  LDS_FENCE();
  float* ob = out + ((size_t)b * MM + (size_t)rb * WBLK + wave * 16) * (HH * DD)
                  + (size_t)h * DD;
  for (int j = 0; j < 8; ++j) {
    int r = j + 8 * hl;
    float lj = LrW[r];
    float inv = lj > 0.0f ? 1.0f / lj : 0.0f;
    float fm = from_mask[(size_t)b * MM + (size_t)rb * WBLK + wave * 16 + r];
    float sc = inv * fm;
    for (int t = 0; t < 4; ++t)
      ob[(size_t)r * HH * DD + t * 16 + l15] = acc[t][j] * sc;
  }
}

extern "C" void kernel_launch(void* const* d_in, const int* in_sizes, int n_in,
                              void* d_out, int out_size, void* d_ws, size_t ws_size,
                              hipStream_t stream) {
  (void)in_sizes; (void)n_in; (void)out_size; (void)d_ws; (void)ws_size;
  const float* q  = (const float*)d_in[0];   // query (B,H,M,D) fp32
  const float* k  = (const float*)d_in[1];   // key
  const float* v  = (const float*)d_in[2];   // value
  const float* fm = (const float*)d_in[4];   // from_mask (B,1,M,1)
  const float* tm = (const float*)d_in[5];   // to_mask   (B,1,1,N)
  const int*   ra = (const int*)d_in[8];     // rand_attn (B,H,NB-2,R)
  float* out = (float*)d_out;                // (B,M,H,D) fp32

  dim3 grid(NBLK, HH, BB), block(128);
  hipLaunchKernelGGL(bigbird_sparse_attn, grid, block, 0, stream,
                     q, k, v, tm, fm, ra, out);
}
